// FeedForward_13503377178799
// MI455X (gfx1250) — matrix operand order repaired and checked by reading, not compiled, not run on hardware
//
#include <hip/hip_runtime.h>
#include <hip/hip_bf16.h>
#include <stdint.h>

// ---------------------------------------------------------------------------
// LoRA SwiGLU FFN for MI455X (gfx1250, wave32, WMMA).
//   1) merge rank-16 LoRA into weights once (W_eff = W + 2*B*A) -> bf16
//   2) GEMM1: g = silu(x*W1e^T) * (x*W3e^T)   (dual-B, shared A tile)
//   3) GEMM2: out = g * W2e^T                 (f32 output)
// Core: v_wmma_f32_16x16x32_bf16, double-buffered LDS fed by
// global_load_async_to_lds_b128 (ASYNCcnt) when available.
// ---------------------------------------------------------------------------

#define DIM   4096
#define HID   11008
#define TOK   8192      // 4 * 2048 tokens
#define RLORA 16
#define LSCALE 2.0f     // lora_alpha / r

typedef __attribute__((ext_vector_type(16))) __bf16 v16bf;
typedef __attribute__((ext_vector_type(8)))  float  v8f;
typedef __attribute__((ext_vector_type(4)))  int    v4i;

__device__ __forceinline__ unsigned short f32_to_bf16_rne(float f) {
    union { float f; uint32_t u; } x; x.f = f;
    uint32_t u = x.u;
    u += 0x7fffu + ((u >> 16) & 1u);   // round to nearest even
    return (unsigned short)(u >> 16);
}

struct BFrag { uint4 lo, hi; };
__device__ __forceinline__ v16bf mkfrag(uint4 lo, uint4 hi) {
    BFrag f; f.lo = lo; f.hi = hi;
    return __builtin_bit_cast(v16bf, f);
}

// --- CDNA5 async global->LDS copy (16B per lane), with safe fallbacks ------
__device__ __forceinline__ void async_copy16(const unsigned short* g,
                                             unsigned short* l) {
#if defined(__AMDGCN__) && __has_builtin(__builtin_amdgcn_global_load_async_to_lds_b128)
    // builtin prototype (from hipcc diagnostic): (v4i AS1*, v4i AS3*, Ii, Ii)
    __builtin_amdgcn_global_load_async_to_lds_b128(
        (__attribute__((address_space(1))) v4i*)(void*)g,
        (__attribute__((address_space(3))) v4i*)(void*)l,
        /*offset=*/0, /*cpol=*/0);
#else
    *(uint4*)l = *(const uint4*)g;
#endif
}

__device__ __forceinline__ void async_wait_all() {
#if defined(__AMDGCN__) && __has_builtin(__builtin_amdgcn_s_wait_asynccnt)
    __builtin_amdgcn_s_wait_asynccnt(0);
#elif defined(__AMDGCN__)
    asm volatile("s_wait_asynccnt 0x0" ::: "memory");
#endif
}

// ---------------------------------------------------------------------------
// Pass 1a: W_eff[o,d] = W[o,d] + LSCALE * sum_r B[o,r]*A[r,d]  -> bf16
// ---------------------------------------------------------------------------
__global__ __launch_bounds__(256)
void merge_lora_kernel(const float* __restrict__ W,
                       const float* __restrict__ A,   // [R, D]
                       const float* __restrict__ B,   // [O, R]
                       unsigned short* __restrict__ Wout, // [O, D] bf16
                       int O, int D) {
    long idx = (long)blockIdx.x * blockDim.x + threadIdx.x;
    long total = (long)O * (long)D;
    if (idx >= total) return;
    int o = (int)(idx / D);
    int d = (int)(idx % D);
    float acc = W[idx];
    #pragma unroll
    for (int r = 0; r < RLORA; ++r)
        acc += LSCALE * B[(long)o * RLORA + r] * A[(long)r * D + d];
    Wout[idx] = f32_to_bf16_rne(acc);
}

// ---------------------------------------------------------------------------
// Pass 1b: x (f32) -> bf16
// ---------------------------------------------------------------------------
__global__ __launch_bounds__(256)
void cvt_bf16_kernel(const float* __restrict__ in,
                     unsigned short* __restrict__ out, long n) {
    long i = (long)blockIdx.x * blockDim.x + threadIdx.x;
    if (i < n) out[i] = f32_to_bf16_rne(in[i]);
}

// ---------------------------------------------------------------------------
// GEMM1: dual-weight fused SwiGLU.
//   A = xb [TOK, DIM] bf16; B = w1e/w3e [HID, DIM] bf16 (both K-contiguous).
// Block: 256 thr = 8 waves; tile 128(M) x 64(N); wave tile 32x32 per weight.
// Double-buffered LDS, async-fed.
// ---------------------------------------------------------------------------
#define G1_BM 128
#define G1_BN 64
#define KT    32
#define LSTR  40   // LDS row stride (bf16) — pads to dodge bank conflicts

__global__ __launch_bounds__(256)
void gemm_swiglu_kernel(const unsigned short* __restrict__ Xb,
                        const unsigned short* __restrict__ W1,
                        const unsigned short* __restrict__ W3,
                        unsigned short* __restrict__ G) {
    __shared__ unsigned short lA [2][G1_BM * LSTR];
    __shared__ unsigned short lB1[2][G1_BN * LSTR];
    __shared__ unsigned short lB3[2][G1_BN * LSTR];

    const int tid   = threadIdx.x;
    const int lane  = tid & 31;
    const int wave  = tid >> 5;
    const int waveM = wave & 3;   // 4 waves along M (32 rows each)
    const int waveN = wave >> 2;  // 2 waves along N (32 cols each)

    const long rowA0 = (long)blockIdx.x * G1_BM;
    const long rowB0 = (long)blockIdx.y * G1_BN;

    // staging decomposition (16B chunks)
    const int rB  = tid >> 2;
    const int ccB = (tid & 3) * 8;

    auto stage = [&](int b, int k0) {
        #pragma unroll
        for (int i = 0; i < 2; ++i) {
            const int c  = tid + i * 256;
            const int r  = c >> 2;
            const int cc = (c & 3) * 8;
            async_copy16(Xb + (rowA0 + r) * DIM + k0 + cc,
                         &lA[b][r * LSTR + cc]);
        }
        async_copy16(W1 + (rowB0 + rB) * DIM + k0 + ccB,
                     &lB1[b][rB * LSTR + ccB]);
        async_copy16(W3 + (rowB0 + rB) * DIM + k0 + ccB,
                     &lB3[b][rB * LSTR + ccB]);
    };

    v8f acc1[2][2], acc3[2][2];
    const v8f vzero = {};
    #pragma unroll
    for (int i = 0; i < 2; ++i)
        #pragma unroll
        for (int j = 0; j < 2; ++j) { acc1[i][j] = vzero; acc3[i][j] = vzero; }

    const int mhalf = lane >> 4;     // which K-half this lane supplies
    const int mrow  = lane & 15;

    // prologue: fill buffer 0
    stage(0, 0);
    async_wait_all();
    __syncthreads();

    const int NT = DIM / KT;
    for (int kt = 0; kt < NT; ++kt) {
        const int buf = kt & 1;
        if (kt + 1 < NT) stage(buf ^ 1, (kt + 1) * KT);   // stream next tile

        // --- fragments from current buffer --------------------------------
        v16bf aF[2], b1F[2], b3F[2];
        #pragma unroll
        for (int i = 0; i < 2; ++i) {
            const unsigned short* p =
                &lA[buf][(waveM * 32 + i * 16 + mrow) * LSTR + mhalf * 16];
            aF[i] = mkfrag(*(const uint4*)p, *(const uint4*)(p + 8));
        }
        #pragma unroll
        for (int j = 0; j < 2; ++j) {
            const unsigned short* p1 =
                &lB1[buf][(waveN * 32 + j * 16 + mrow) * LSTR + mhalf * 16];
            b1F[j] = mkfrag(*(const uint4*)p1, *(const uint4*)(p1 + 8));
            const unsigned short* p3 =
                &lB3[buf][(waveN * 32 + j * 16 + mrow) * LSTR + mhalf * 16];
            b3F[j] = mkfrag(*(const uint4*)p3, *(const uint4*)(p3 + 8));
        }
        // --- 8 WMMAs per K step -------------------------------------------
        #pragma unroll
        for (int i = 0; i < 2; ++i)
            #pragma unroll
            for (int j = 0; j < 2; ++j) {
                acc1[i][j] = __builtin_amdgcn_wmma_f32_16x16x32_bf16(
                    false, aF[i], false, b1F[j], (short)0, acc1[i][j], false, false);
                acc3[i][j] = __builtin_amdgcn_wmma_f32_16x16x32_bf16(
                    false, aF[i], false, b3F[j], (short)0, acc3[i][j], false, false);
            }
        async_wait_all();     // next buffer landed
        __syncthreads();      // everyone done reading current buffer
    }

    // --- epilogue: g = silu(h1) * h3 -> bf16 ------------------------------
    const int n_lane = lane & 15;
    const int m_off  = (lane >> 4) * 8;
    #pragma unroll
    for (int i = 0; i < 2; ++i)
        #pragma unroll
        for (int j = 0; j < 2; ++j)
            #pragma unroll
            for (int v = 0; v < 8; ++v) {
                const long row = rowA0 + waveM * 32 + i * 16 + m_off + v;
                const long col = rowB0 + waveN * 32 + j * 16 + n_lane;
                const float h1 = acc1[i][j][v];
                const float h3 = acc3[i][j][v];
                const float g  = (h1 / (1.0f + __expf(-h1))) * h3;
                G[row * HID + col] = f32_to_bf16_rne(g);
            }
}

// ---------------------------------------------------------------------------
// GEMM2: out[t,d] = sum_h g[t,h] * w2e[d,h]   (f32 out)
// Block tile 128x128; 8 waves as 2(M)x4(N); wave tile 64x32.
// ---------------------------------------------------------------------------
#define G2_BM 128
#define G2_BN 128

__global__ __launch_bounds__(256)
void gemm_out_kernel(const unsigned short* __restrict__ Gb,  // [TOK, HID] bf16
                     const unsigned short* __restrict__ W2,  // [DIM, HID] bf16
                     float* __restrict__ Out) {              // [TOK, DIM] f32
    __shared__ unsigned short lA[2][G2_BM * LSTR];
    __shared__ unsigned short lB[2][G2_BN * LSTR];

    const int tid   = threadIdx.x;
    const int lane  = tid & 31;
    const int wave  = tid >> 5;
    const int waveM = wave >> 2;  // 2 waves along M (64 rows each)
    const int waveN = wave & 3;   // 4 waves along N (32 cols each)

    const long rowA0 = (long)blockIdx.x * G2_BM;
    const long rowB0 = (long)blockIdx.y * G2_BN;

    auto stage = [&](int b, int k0) {
        #pragma unroll
        for (int i = 0; i < 2; ++i) {
            const int c  = tid + i * 256;
            const int r  = c >> 2;
            const int cc = (c & 3) * 8;
            async_copy16(Gb + (rowA0 + r) * HID + k0 + cc,
                         &lA[b][r * LSTR + cc]);
            async_copy16(W2 + (rowB0 + r) * HID + k0 + cc,
                         &lB[b][r * LSTR + cc]);
        }
    };

    v8f acc[4][2];
    const v8f vzero = {};
    #pragma unroll
    for (int i = 0; i < 4; ++i)
        #pragma unroll
        for (int j = 0; j < 2; ++j) acc[i][j] = vzero;

    const int mhalf = lane >> 4;
    const int mrow  = lane & 15;

    stage(0, 0);
    async_wait_all();
    __syncthreads();

    const int NT = HID / KT;
    for (int kt = 0; kt < NT; ++kt) {
        const int buf = kt & 1;
        if (kt + 1 < NT) stage(buf ^ 1, (kt + 1) * KT);

        v16bf aF[4], bF[2];
        #pragma unroll
        for (int i = 0; i < 4; ++i) {
            const unsigned short* p =
                &lA[buf][(waveM * 64 + i * 16 + mrow) * LSTR + mhalf * 16];
            aF[i] = mkfrag(*(const uint4*)p, *(const uint4*)(p + 8));
        }
        #pragma unroll
        for (int j = 0; j < 2; ++j) {
            const unsigned short* p =
                &lB[buf][(waveN * 32 + j * 16 + mrow) * LSTR + mhalf * 16];
            bF[j] = mkfrag(*(const uint4*)p, *(const uint4*)(p + 8));
        }
        #pragma unroll
        for (int i = 0; i < 4; ++i)
            #pragma unroll
            for (int j = 0; j < 2; ++j)
                acc[i][j] = __builtin_amdgcn_wmma_f32_16x16x32_bf16(
                    false, aF[i], false, bF[j], (short)0, acc[i][j], false, false);
        async_wait_all();
        __syncthreads();
    }

    const int n_lane = lane & 15;
    const int m_off  = (lane >> 4) * 8;
    #pragma unroll
    for (int i = 0; i < 4; ++i)
        #pragma unroll
        for (int j = 0; j < 2; ++j)
            #pragma unroll
            for (int v = 0; v < 8; ++v) {
                const long row = rowA0 + waveM * 64 + i * 16 + m_off + v;
                const long col = rowB0 + waveN * 32 + j * 16 + n_lane;
                Out[row * DIM + col] = acc[i][j][v];
            }
}

// ---------------------------------------------------------------------------
// Host launch
// ---------------------------------------------------------------------------
extern "C" void kernel_launch(void* const* d_in, const int* in_sizes, int n_in,
                              void* d_out, int out_size, void* d_ws, size_t ws_size,
                              hipStream_t stream) {
    const float* x  = (const float*)d_in[0];
    const float* w1 = (const float*)d_in[1];
    const float* w3 = (const float*)d_in[2];
    const float* w2 = (const float*)d_in[3];
    const float* A1 = (const float*)d_in[4];
    const float* B1 = (const float*)d_in[5];
    const float* A3 = (const float*)d_in[6];
    const float* B3 = (const float*)d_in[7];
    const float* A2 = (const float*)d_in[8];
    const float* B2 = (const float*)d_in[9];
    float* out = (float*)d_out;

    // workspace layout (bf16 buffers)
    char* ws = (char*)d_ws;
    size_t off = 0;
    auto alloc = [&](size_t bytes) {
        char* p = ws + off;
        off = (off + bytes + 255) & ~(size_t)255;
        return p;
    };
    unsigned short* xb  = (unsigned short*)alloc((size_t)TOK * DIM * 2);
    unsigned short* w1e = (unsigned short*)alloc((size_t)HID * DIM * 2);
    unsigned short* w3e = (unsigned short*)alloc((size_t)HID * DIM * 2);
    unsigned short* w2e = (unsigned short*)alloc((size_t)DIM * HID * 2);
    unsigned short* g   = (unsigned short*)alloc((size_t)TOK * HID * 2);
    (void)ws_size; (void)n_in; (void)in_sizes; (void)out_size;

    // Pass 1: merge LoRA into weights (bf16) + convert activations
    {
        long n = (long)HID * DIM;
        int blocks = (int)((n + 255) / 256);
        merge_lora_kernel<<<blocks, 256, 0, stream>>>(w1, A1, B1, w1e, HID, DIM);
        merge_lora_kernel<<<blocks, 256, 0, stream>>>(w3, A3, B3, w3e, HID, DIM);
        merge_lora_kernel<<<blocks, 256, 0, stream>>>(w2, A2, B2, w2e, DIM, HID);
    }
    {
        long n = (long)TOK * DIM;
        cvt_bf16_kernel<<<(int)((n + 255) / 256), 256, 0, stream>>>(x, xb, n);
    }

    // Pass 2: fused SwiGLU GEMM  (g = silu(x*W1e^T) * (x*W3e^T))
    {
        dim3 grid(TOK / G1_BM, HID / G1_BN);   // 64 x 172
        gemm_swiglu_kernel<<<grid, 256, 0, stream>>>(xb, w1e, w3e, g);
    }
    // Pass 3: output GEMM (out = g * W2e^T)
    {
        dim3 grid(TOK / G2_BM, DIM / G2_BN);   // 64 x 32
        gemm_out_kernel<<<grid, 256, 0, stream>>>(g, w2e, out);
    }
}
